// FullHoloLayer_7851200217998
// MI455X (gfx1250) — matrix operand order repaired and checked by reading, not compiled, not run on hardware
//
#include <hip/hip_runtime.h>

// ---------------------------------------------------------------------------
// FullHoloLayer on MI455X (gfx1250): bf16 WMMA GEMMs (async-LDS double-
// buffered) + LDS Stockham FFTs + chunked causal scan.
// B=2, S=2048, D=2048. ~725 GFLOP of matmul via v_wmma_f32_16x16x32_bf16.
// ---------------------------------------------------------------------------

#define Bdim 2
#define Sdim 2048
#define Ddim 2048
#define NROWS (Bdim * Sdim)          // 4096
#define BD (Bdim * Ddim)             // 4096 scan channels
#define SCHUNK 128                   // scan chunk length
#define NCHUNK (Sdim / SCHUNK)       // 16

typedef __attribute__((ext_vector_type(16))) __bf16 bf16x16;
typedef __attribute__((ext_vector_type(8)))  float  f32x8;
typedef __attribute__((ext_vector_type(4)))  int    v4i;

#define AS1 __attribute__((address_space(1)))
#define AS3 __attribute__((address_space(3)))

#if __has_builtin(__builtin_amdgcn_global_load_async_to_lds_b128)
#define HAVE_ASYNC_LDS 1
#else
#define HAVE_ASYNC_LDS 0
#endif

__device__ __forceinline__ __bf16 f2bf(float f) {
  union { float f; unsigned u; } v; v.f = f;
  unsigned r = (v.u + 0x7FFFu + ((v.u >> 16) & 1u)) >> 16;
  union { unsigned short s; __bf16 b; } o; o.s = (unsigned short)r;
  return o.b;
}
__device__ __forceinline__ float2 cmul(float2 a, float2 b) {
  return make_float2(a.x * b.x - a.y * b.y, a.x * b.y + a.y * b.x);
}

template <int N>
__device__ __forceinline__ void wait_asynccnt() {
#if __has_builtin(__builtin_amdgcn_s_wait_asynccnt)
  __builtin_amdgcn_s_wait_asynccnt(N);
#else
  asm volatile("s_wait_asynccnt %0" :: "i"(N) : "memory");
#endif
}

#if HAVE_ASYNC_LDS
__device__ __forceinline__ void async_copy_b128(const void* g, void* l) {
  // global -> LDS, 16 bytes per lane, tracked by ASYNCcnt.
  // Param 0: int4 ptr in addrspace(1); param 1: int4 ptr in addrspace(3).
  __builtin_amdgcn_global_load_async_to_lds_b128((AS1 v4i*)g, (AS3 v4i*)l, 0, 0);
}
#endif

// ---------------------------------------------------------------------------
// f32 -> bf16 cast
// ---------------------------------------------------------------------------
__global__ void cast_f32_bf16(const float* __restrict__ in,
                              __bf16* __restrict__ out, int n) {
  int i = (blockIdx.x * blockDim.x + threadIdx.x) * 4;
  if (i + 3 < n) {
    float4 v = *(const float4*)(in + i);
    out[i + 0] = f2bf(v.x);
    out[i + 1] = f2bf(v.y);
    out[i + 2] = f2bf(v.z);
    out[i + 3] = f2bf(v.w);
  } else {
    for (; i < n; ++i) out[i] = f2bf(in[i]);
  }
}

// ---------------------------------------------------------------------------
// bf16 WMMA GEMM: C[M,N] = A[M,K] @ B[K,N] (+bias, optional relu)
// WG = 256 threads = 8 waves; block tile 128x128, K-step 32.
// Waves 4x2; each wave owns 32x64 = 2x4 WMMA 16x16 tiles (8 WMMA / K-step).
// Double-buffered LDS staged with GLOBAL_LOAD_ASYNC_TO_LDS_B128 when
// available (ASYNCcnt pipeline), else sync loads.
// A frag: lane%16 = M, lane/16 = K half-block, VGPR pair = K pair.
// B frag: lane%16 = N, lane/16 = K half-block, VGPR pair = K pair (same k order as A).
// D frag: n = lane%16, m = vgpr + 8*(lane/16).
// ---------------------------------------------------------------------------
#define BM 128
#define BN 128
#define BK 32

template <bool HAS_BIAS, bool RELU, bool OUT_F32>
__global__ __launch_bounds__(256)
void gemm_bf16_wmma(const __bf16* __restrict__ A, const __bf16* __restrict__ B,
                    float* __restrict__ Cf, __bf16* __restrict__ Cb,
                    const float* __restrict__ bias, int M, int N, int K) {
  __shared__ alignas(16) __bf16 As[2][BM][BK];   // 2 x 8 KB
  __shared__ alignas(16) __bf16 Bs[2][BK][BN];   // 2 x 8 KB
  const int tid  = threadIdx.x;
  const int wave = tid >> 5;
  const int lane = tid & 31;
  const int wm = wave >> 1;        // 0..3 -> 32-row M sub-tile
  const int wn = wave & 1;         // 0..1 -> 64-col N sub-tile
  const int mL = lane & 15;
  const int g  = lane >> 4;
  const int rowBase = blockIdx.y * BM;
  const int colBase = blockIdx.x * BN;

  // staging map: each thread moves 32B of A and 32B of B per K-tile
  const int ar = tid >> 1;              // A row 0..127
  const int ac = (tid & 1) * 16;        // A col chunk {0,16}
  const int bk = tid >> 3;              // B row (k) 0..31
  const int bn = (tid & 7) * 16;        // B col chunk 0..112

  const __bf16* gA = A + (size_t)(rowBase + ar) * K + ac;
  const __bf16* gB = B + (size_t)bk * N + colBase + bn;

  f32x8 acc[2][4] = {};
  const int nK = K / BK;

#if HAVE_ASYNC_LDS
  // prologue: stage tile 0 into buffer 0
  async_copy_b128(gA,     &As[0][ar][ac]);
  async_copy_b128(gA + 8, &As[0][ar][ac + 8]);
  async_copy_b128(gB,     &Bs[0][bk][bn]);
  async_copy_b128(gB + 8, &Bs[0][bk][bn + 8]);
#endif

  for (int kt = 0; kt < nK; ++kt) {
    const int cur = kt & 1;
#if HAVE_ASYNC_LDS
    if (kt + 1 < nK) {
      const int nxt = (kt + 1) & 1;
      const size_t ka = (size_t)(kt + 1) * BK;
      async_copy_b128(gA + ka,            &As[nxt][ar][ac]);
      async_copy_b128(gA + ka + 8,        &As[nxt][ar][ac + 8]);
      async_copy_b128(gB + ka * N,        &Bs[nxt][bk][bn]);
      async_copy_b128(gB + ka * N + 8,    &Bs[nxt][bk][bn + 8]);
      wait_asynccnt<4>();    // tile kt complete; tile kt+1 in flight
    } else {
      wait_asynccnt<0>();
    }
#else
    {
      const size_t ka = (size_t)kt * BK;
      const float4* a4 = (const float4*)(gA + ka);
      const float4* b4 = (const float4*)(gB + ka * N);
      float4 va0 = a4[0], va1 = a4[1];
      float4 vb0 = b4[0], vb1 = b4[1];
      *(float4*)&As[cur][ar][ac + 0] = va0;
      *(float4*)&As[cur][ar][ac + 8] = va1;
      *(float4*)&Bs[cur][bk][bn + 0] = vb0;
      *(float4*)&Bs[cur][bk][bn + 8] = vb1;
      __builtin_prefetch(gA + ka + BK, 0, 1);
    }
#endif
    __syncthreads();

    // --- fragments ---
    bf16x16 af[2], bfr[4];
#pragma unroll
    for (int mt = 0; mt < 2; ++mt) {
      const int row = wm * 32 + mt * 16 + mL;
#pragma unroll
      for (int r = 0; r < 8; ++r) {
        const int kb = (r < 4) ? (g * 8 + r * 2) : (16 + g * 8 + (r - 4) * 2);
        af[mt][2 * r + 0] = As[cur][row][kb + 0];
        af[mt][2 * r + 1] = As[cur][row][kb + 1];
      }
    }
#pragma unroll
    for (int nt = 0; nt < 4; ++nt) {
      const int nb = wn * 64 + nt * 16;
#pragma unroll
      for (int r = 0; r < 8; ++r) {
        const int kb = (r < 4) ? (g * 8 + r * 2) : (16 + g * 8 + (r - 4) * 2);
        bfr[nt][2 * r + 0] = Bs[cur][kb + 0][nb + mL];
        bfr[nt][2 * r + 1] = Bs[cur][kb + 1][nb + mL];
      }
    }

    // --- 8 WMMAs per K-step ---
#pragma unroll
    for (int mt = 0; mt < 2; ++mt)
#pragma unroll
      for (int nt = 0; nt < 4; ++nt)
        acc[mt][nt] = __builtin_amdgcn_wmma_f32_16x16x32_bf16(
            false, af[mt], false, bfr[nt], (short)0, acc[mt][nt], false, false);
    __syncthreads();
  }

  // --- branch-free epilogue (compile-time configuration) ---
#pragma unroll
  for (int mt = 0; mt < 2; ++mt)
#pragma unroll
    for (int nt = 0; nt < 4; ++nt)
#pragma unroll
      for (int r = 0; r < 8; ++r) {
        const int row = rowBase + wm * 32 + mt * 16 + r + 8 * g;
        const int col = colBase + wn * 64 + nt * 16 + mL;
        float v = acc[mt][nt][r];
        if constexpr (HAS_BIAS) v += bias[col];
        if constexpr (RELU) v = v > 0.f ? v : 0.f;
        const size_t idx = (size_t)row * N + col;
        if constexpr (OUT_F32) Cf[idx] = v;
        else                   Cb[idx] = f2bf(v);
      }
}

// ---------------------------------------------------------------------------
// 2048-point Stockham radix-2 FFT in LDS, 256 threads, 4 butterflies/thread.
// ---------------------------------------------------------------------------
__device__ float2* fft2048(float2* b0, float2* b1, int tid, float sign) {
  float2* src = b0;
  float2* dst = b1;
#pragma unroll 1
  for (int s = 0; s < 11; ++s) {
    const int mh = 1 << s;
    __syncthreads();
    for (int t = tid; t < 1024; t += 256) {
      const int k   = t & (mh - 1);
      const int blk = t >> s;
      const float ang = sign * 6.28318530717958647692f * (float)k / (float)(2 * mh);
      const float c = __cosf(ang), sn = __sinf(ang);
      const float2 a = src[blk * mh + k];
      const float2 b = src[blk * mh + k + 1024];
      const float2 wb = make_float2(c * b.x - sn * b.y, c * b.y + sn * b.x);
      dst[blk * (2 * mh) + k]      = make_float2(a.x + wb.x, a.y + wb.y);
      dst[blk * (2 * mh) + k + mh] = make_float2(a.x - wb.x, a.y - wb.y);
    }
    float2* tmp = src; src = dst; dst = tmp;
  }
  __syncthreads();
  return src;
}

// ---------------------------------------------------------------------------
// Per-row forward stage: FFT x,q,k,v,s,t; attn = conj(Fq)*Fk*Fv (pre-scan).
// LDS: ping+pong + Fq + Fk = 64KB (<< 320KB per WGP).
// ---------------------------------------------------------------------------
__global__ __launch_bounds__(256)
void fft_forward_kernel(const float* __restrict__ x,  const float* __restrict__ q,
                        const float* __restrict__ kp, const float* __restrict__ vp,
                        const float* __restrict__ sp, const float* __restrict__ tp,
                        float2* __restrict__ Xhat, float2* __restrict__ Attn,
                        float2* __restrict__ Fs,   float2* __restrict__ Ft) {
  __shared__ float2 b0[Ddim];
  __shared__ float2 b1[Ddim];
  __shared__ float2 Qb[Ddim];
  __shared__ float2 Kb[Ddim];
  const int row = blockIdx.x;
  const int tid = threadIdx.x;
  const size_t off = (size_t)row * Ddim;

  for (int i = tid; i < Ddim; i += 256) b0[i] = make_float2(x[off + i], 0.f);
  float2* r = fft2048(b0, b1, tid, -1.f);
  for (int i = tid; i < Ddim; i += 256) Xhat[off + i] = r[i];
  __syncthreads();

  for (int i = tid; i < Ddim; i += 256) b0[i] = make_float2(q[off + i], 0.f);
  r = fft2048(b0, b1, tid, -1.f);
  for (int i = tid; i < Ddim; i += 256) Qb[i] = r[i];
  __syncthreads();

  for (int i = tid; i < Ddim; i += 256) b0[i] = make_float2(kp[off + i], 0.f);
  r = fft2048(b0, b1, tid, -1.f);
  for (int i = tid; i < Ddim; i += 256) Kb[i] = r[i];
  __syncthreads();

  for (int i = tid; i < Ddim; i += 256) b0[i] = make_float2(vp[off + i], 0.f);
  r = fft2048(b0, b1, tid, -1.f);
  for (int i = tid; i < Ddim; i += 256) {
    const float2 qc = make_float2(Qb[i].x, -Qb[i].y);
    Attn[off + i] = cmul(cmul(qc, Kb[i]), r[i]);
  }
  __syncthreads();

  for (int i = tid; i < Ddim; i += 256) b0[i] = make_float2(sp[off + i], 0.f);
  r = fft2048(b0, b1, tid, -1.f);
  for (int i = tid; i < Ddim; i += 256) Fs[off + i] = r[i];
  __syncthreads();

  for (int i = tid; i < Ddim; i += 256) b0[i] = make_float2(tp[off + i], 0.f);
  r = fft2048(b0, b1, tid, -1.f);
  for (int i = tid; i < Ddim; i += 256) Ft[off + i] = r[i];
}

// ---------------------------------------------------------------------------
// Chunked causal cumsum over S (3 passes, bandwidth-bound):
//  pass1: per (chunk, b, d) partial sums of Attn            (65536 threads)
//  pass2: exclusive scan of NCHUNK partials per channel     (4096 threads)
//  pass3: apply within chunk; fuse x_hat update, t_args, tr (65536 threads)
// ---------------------------------------------------------------------------
__global__ void scan_pass1(const float2* __restrict__ Attn, float2* __restrict__ Part) {
  const int t = blockIdx.x * blockDim.x + threadIdx.x;   // 0..NCHUNK*BD-1
  const int cd = t & (BD - 1);
  const int chunk = t >> 12;                             // /BD
  const int b = cd >> 11;
  const int d = cd & (Ddim - 1);
  float2 acc = make_float2(0.f, 0.f);
  const int s0 = chunk * SCHUNK;
  for (int s = s0; s < s0 + SCHUNK; ++s) {
    const float2 a = Attn[((size_t)b * Sdim + s) * Ddim + d];
    acc.x += a.x; acc.y += a.y;
  }
  Part[(size_t)chunk * BD + cd] = acc;
}

__global__ void scan_pass2(float2* __restrict__ Part) {
  const int cd = blockIdx.x * blockDim.x + threadIdx.x;  // 0..BD-1
  float2 run = make_float2(0.f, 0.f);
  for (int c = 0; c < NCHUNK; ++c) {
    const size_t i = (size_t)c * BD + cd;
    const float2 v = Part[i];
    Part[i] = run;                 // exclusive prefix
    run.x += v.x; run.y += v.y;
  }
}

__global__ void scan_pass3(float2* __restrict__ Xhat, const float2* __restrict__ Attn,
                           const float2* __restrict__ Fs, const float2* __restrict__ Part,
                           float2* __restrict__ Targs, __bf16* __restrict__ tr) {
  const int t = blockIdx.x * blockDim.x + threadIdx.x;
  const int cd = t & (BD - 1);
  const int chunk = t >> 12;
  const int b = cd >> 11;
  const int d = cd & (Ddim - 1);
  float2 acc = Part[(size_t)chunk * BD + cd];
  const int s0 = chunk * SCHUNK;
  for (int s = s0; s < s0 + SCHUNK; ++s) {
    const size_t idx = ((size_t)b * Sdim + s) * Ddim + d;
    const float2 a = Attn[idx];
    acc.x += a.x; acc.y += a.y;
    float2 xh = Xhat[idx];
    xh.x += acc.x; xh.y += acc.y;
    Xhat[idx] = xh;
    const float2 fs = Fs[idx];
    const float2 ta = cmul(make_float2(fs.x, -fs.y), xh);
    Targs[idx] = ta;
    const size_t trow = ((size_t)b * Sdim + s) * (2 * Ddim);
    tr[trow + d]        = f2bf(ta.x);
    tr[trow + Ddim + d] = f2bf(ta.y);
  }
}

// ---------------------------------------------------------------------------
// Final per-row: x_hat + Ft*complex(h) - Fs*t_args, inverse FFT, take real.
// ---------------------------------------------------------------------------
__global__ __launch_bounds__(256)
void final_kernel(const float* __restrict__ h, const float2* __restrict__ Ft,
                  const float2* __restrict__ Fs, const float2* __restrict__ Targs,
                  const float2* __restrict__ Xhat, float* __restrict__ out) {
  __shared__ float2 b0[Ddim];
  __shared__ float2 b1[Ddim];
  const int row = blockIdx.x;
  const int tid = threadIdx.x;
  const size_t off  = (size_t)row * Ddim;
  const size_t hoff = (size_t)row * (2 * Ddim);
  for (int i = tid; i < Ddim; i += 256) {
    const float2 tc = make_float2(h[hoff + i], h[hoff + Ddim + i]);
    const float2 add = cmul(Ft[off + i], tc);
    const float2 sub = cmul(Fs[off + i], Targs[off + i]);
    const float2 xh = Xhat[off + i];
    b0[i] = make_float2(xh.x + add.x - sub.x, xh.y + add.y - sub.y);
  }
  float2* r = fft2048(b0, b1, tid, +1.f);
  const float invn = 1.0f / (float)Ddim;
  for (int i = tid; i < Ddim; i += 256) out[off + i] = r[i].x * invn;
}

// ---------------------------------------------------------------------------
// Launch: cast -> 5 proj GEMMs -> FFT stage -> scan x3 -> MLP GEMMs -> iFFT.
// ---------------------------------------------------------------------------
extern "C" void kernel_launch(void* const* d_in, const int* in_sizes, int n_in,
                              void* d_out, int out_size, void* d_ws, size_t ws_size,
                              hipStream_t stream) {
  const float* x  = (const float*)d_in[0];
  const float* Wf[5] = { (const float*)d_in[1], (const float*)d_in[2],
                         (const float*)d_in[3], (const float*)d_in[4],
                         (const float*)d_in[5] };
  const float* W1 = (const float*)d_in[6];
  const float* b1 = (const float*)d_in[7];
  const float* W2 = (const float*)d_in[8];
  const float* b2 = (const float*)d_in[9];
  float* out = (float*)d_out;

  const size_t MB = 1ull << 20;
  char* w = (char*)d_ws;
  __bf16* xb   = (__bf16*)(w + 0);                    // 16 MB
  __bf16* Wb   = (__bf16*)(w + 16 * MB);              // 5 x 8 MB
  __bf16* W1b  = (__bf16*)(w + 56 * MB);              // 64 MB
  __bf16* W2b  = (__bf16*)(w + 120 * MB);             // 64 MB
  float*  P    = (float*)(w + 184 * MB);              // 5 x 32 MB (q,k,v,s,t)
  float2* Xhat = (float2*)(w + 344 * MB);             // 64 MB
  float2* Attn = (float2*)(w + 408 * MB);             // 64 MB
  float2* Fs   = (float2*)(w + 472 * MB);             // 64 MB
  float2* Ft   = (float2*)(w + 536 * MB);             // 64 MB
  float2* Ta   = (float2*)(w + 600 * MB);             // 64 MB
  float2* Part = (float2*)(w + 664 * MB);             // 512 KB scan partials
  // aliases over dead projection region:
  __bf16* tr  = (__bf16*)(w + 184 * MB);              // 32 MB
  __bf16* h1  = (__bf16*)(w + 184 * MB + 32 * MB);    // 64 MB
  float*  h   = (float*)(w + 184 * MB + 96 * MB);     // 64 MB
  (void)ws_size; (void)n_in; (void)in_sizes; (void)out_size;

  auto cast = [&](const float* src, __bf16* dst, int n) {
    cast_f32_bf16<<<(n / 4 + 255) / 256, 256, 0, stream>>>(src, dst, n);
  };
  const int nX  = NROWS * Ddim;          // 8388608
  const int nW  = Ddim * Ddim;           // 4194304
  const int nW1 = 2 * Ddim * 4 * Ddim;   // 33554432
  cast(x, xb, nX);
  for (int i = 0; i < 5; ++i) cast(Wf[i], Wb + (size_t)i * nW, nW);
  cast(W1, W1b, nW1);
  cast(W2, W2b, nW1);

  // 5 projection GEMMs: [4096,2048] = [4096,2048] @ [2048,2048]
  for (int i = 0; i < 5; ++i) {
    gemm_bf16_wmma<false, false, true><<<dim3(Ddim / BN, NROWS / BM), 256, 0, stream>>>(
        xb, Wb + (size_t)i * nW, P + (size_t)i * nX, nullptr, nullptr,
        NROWS, Ddim, Ddim);
  }

  // FFT forward stage (6 FFTs per row, fused attn product)
  fft_forward_kernel<<<NROWS, 256, 0, stream>>>(
      x, P + 0 * (size_t)nX, P + 1 * (size_t)nX, P + 2 * (size_t)nX,
      P + 3 * (size_t)nX, P + 4 * (size_t)nX, Xhat, Attn, Fs, Ft);

  // chunked causal cumsum + t_args + tr packing
  scan_pass1<<<(NCHUNK * BD) / 256, 256, 0, stream>>>(Attn, Part);
  scan_pass2<<<BD / 256, 256, 0, stream>>>(Part);
  scan_pass3<<<(NCHUNK * BD) / 256, 256, 0, stream>>>(Xhat, Attn, Fs, Part, Ta, tr);

  // MLP: [4096,8192] = relu([4096,4096] @ [4096,8192] + b1)  (bf16 out)
  gemm_bf16_wmma<true, true, false><<<dim3(4 * Ddim / BN, NROWS / BM), 256, 0, stream>>>(
      tr, W1b, nullptr, h1, b1, NROWS, 4 * Ddim, 2 * Ddim);
  // [4096,4096] = [4096,8192] @ [8192,4096] + b2  (f32 out)
  gemm_bf16_wmma<true, false, true><<<dim3(2 * Ddim / BN, NROWS / BM), 256, 0, stream>>>(
      h1, W2b, h, nullptr, b2, NROWS, 2 * Ddim, 4 * Ddim);

  // recombine + inverse FFT + real part
  final_kernel<<<NROWS, 256, 0, stream>>>(h, Ft, Fs, Ta, Xhat, out);
}
